// Attention_85031762526797
// MI455X (gfx1250) — compile-verified, matrix operand order checked by reading, not run on hardware
//
#include <hip/hip_runtime.h>

// ---------------------------------------------------------------------------
// MI455X (gfx1250) attention block.
// All matmuls on v_wmma_f32_16x16x32_f16; all tile staging via
// global_load_async_to_lds_b128 (ASYNCcnt) with ping-pong LDS buffers in the GEMM.
// ---------------------------------------------------------------------------

typedef __attribute__((ext_vector_type(16))) _Float16 v16h;
typedef __attribute__((ext_vector_type(8)))  float    v8f;

static constexpr int B_   = 2;
static constexpr int L_   = 2048;
static constexpr int DIM_ = 2048;
static constexpr int H_   = 16;
static constexpr int D_   = 128;
static constexpr float EPS_   = 1e-6f;
static constexpr float CAP_   = 50.0f;
static constexpr float SCALE_ = 0.08838834764831845f; // 1/sqrt(128)

// K-index inside a 16-bit A/B fragment of v_wmma_*_16x16x32:
// lanes 0-15 hold K 0..7 (vgpr0-3) and 16..23 (vgpr4-7);
// lanes 16-31 hold K 8..15 and 24..31.  (CDNA5 ISA layout tables)
__device__ __forceinline__ int kmap16(int j, int hi) {
  return (j < 8) ? (hi * 8 + j) : (16 + hi * 8 + (j - 8));
}

// Low 32 bits of a flat shared-memory address are the LDS byte offset
// (ISA 10.2, LDS aperture).
__device__ __forceinline__ unsigned lds_off(const void* p) {
  return (unsigned)(unsigned long long)p;
}

// Async 16-byte global -> LDS copy (per lane), tracked by ASYNCcnt.
__device__ __forceinline__ void async_copy_b128(unsigned lds_addr, const void* g) {
  asm volatile("global_load_async_to_lds_b128 %0, %1, off"
               :: "v"(lds_addr), "v"((unsigned long long)g) : "memory");
}
__device__ __forceinline__ void wait_asynccnt0() {
  asm volatile("s_wait_asynccnt 0" ::: "memory");
}

__device__ __forceinline__ v8f wmma_f16(v16h a, v16h b, v8f c) {
  return __builtin_amdgcn_wmma_f32_16x16x32_f16(false, a, false, b,
                                                (short)0, c, false, false);
}

// ---------------------------------------------------------------------------
// Elementwise f32 -> f16 conversion (n multiple of 2048)
// ---------------------------------------------------------------------------
__global__ __launch_bounds__(256) void cvt_f32_f16(
    const float* __restrict__ in, _Float16* __restrict__ out, int n)
{
  const int i = (blockIdx.x * 256 + threadIdx.x) * 8;
  if (i + 8 <= n) {
#pragma unroll
    for (int j = 0; j < 8; ++j) out[i + j] = (_Float16)in[i + j];
  }
}

// ---------------------------------------------------------------------------
// C[M,N] = A[M,K] * Bw[N,K]^T   (f16 operands, f32 accumulate/output)
// BM=128, BN=64, BK=32; 256 threads = 8 waves; wave computes 2x2 16x16 tiles.
// Async double-buffered LDS staging (3 async b128 per thread per K-slice).
// ---------------------------------------------------------------------------
__global__ __launch_bounds__(256) void gemm_h(
    const _Float16* __restrict__ A, const _Float16* __restrict__ Bw,
    float* __restrict__ C, int M, int N, int K)
{
  constexpr int LDT = 40; // 32 + 8 halves padding; 80 B row stride
  __shared__ __align__(16) _Float16 As[2][128 * LDT];
  __shared__ __align__(16) _Float16 Bs[2][64 * LDT];

  const int tid  = threadIdx.x;
  const int lane = tid & 31;
  const int w    = tid >> 5;
  const int r16  = lane & 15;
  const int hi   = lane >> 4;

  const int m0 = blockIdx.y * 128;
  const int n0 = blockIdx.x * 64;

  const int mt0 = (w & 3) * 2;  // m-tiles {mt0, mt0+1} of 8
  const int nt0 = (w >> 2) * 2; // n-tiles {nt0, nt0+1} of 4

  // staging geometry (halves): A row 0..127, 2 chunks; B row 0..63, 1 chunk
  const int ar = tid >> 1, ac = (tid & 1) * 16;
  const int br = tid >> 2, bc = (tid & 3) * 8;

  v8f acc[2][2] = {};

  auto stage = [&](int buf, int k0) {
    const _Float16* ap = A  + (size_t)(m0 + ar) * K + k0 + ac;
    const _Float16* bp = Bw + (size_t)(n0 + br) * K + k0 + bc;
    const unsigned ad = lds_off(&As[buf][ar * LDT + ac]);
    const unsigned bd = lds_off(&Bs[buf][br * LDT + bc]);
    async_copy_b128(ad,      ap);
    async_copy_b128(ad + 16, ap + 8);
    async_copy_b128(bd,      bp);
  };

  stage(0, 0);
  wait_asynccnt0();
  __syncthreads();

  int buf = 0;
  for (int k0 = 0; k0 < K; k0 += 32) {
    if (k0 + 32 < K) stage(buf ^ 1, k0 + 32); // overlap with compute

    v16h a0, a1, b0, b1;
#pragma unroll
    for (int j = 0; j < 16; ++j) {
      const int kk = kmap16(j, hi);
      a0[j] = As[buf][((mt0 + 0) * 16 + r16) * LDT + kk];
      a1[j] = As[buf][((mt0 + 1) * 16 + r16) * LDT + kk];
      b0[j] = Bs[buf][((nt0 + 0) * 16 + r16) * LDT + kk];
      b1[j] = Bs[buf][((nt0 + 1) * 16 + r16) * LDT + kk];
    }
    acc[0][0] = wmma_f16(a0, b0, acc[0][0]);
    acc[0][1] = wmma_f16(a0, b1, acc[0][1]);
    acc[1][0] = wmma_f16(a1, b0, acc[1][0]);
    acc[1][1] = wmma_f16(a1, b1, acc[1][1]);

    wait_asynccnt0();
    __syncthreads();
    buf ^= 1;
  }

  // D layout: element r -> M = (lane>>4)*8 + r, N = lane&15
#pragma unroll
  for (int i = 0; i < 2; ++i)
#pragma unroll
    for (int jn = 0; jn < 2; ++jn)
#pragma unroll
      for (int r = 0; r < 8; ++r) {
        const int row = m0 + (mt0 + i) * 16 + hi * 8 + r;
        const int col = n0 + (nt0 + jn) * 16 + r16;
        C[(size_t)row * N + col] = acc[i][jn][r];
      }
}

// ---------------------------------------------------------------------------
// RMS-norm + RoPE. One 128-thread block per (b,h,l) row.
// f32 K/V -> d_out sections (kernel outputs);
// f16 Q/K/V -> workspace for the WMMA attention kernel.
// ---------------------------------------------------------------------------
__global__ __launch_bounds__(128) void normrope(
    const float* __restrict__ qkv, const float* __restrict__ qw,
    const float* __restrict__ kw, const float* __restrict__ cosT,
    const float* __restrict__ sinT,
    _Float16* __restrict__ Qh, _Float16* __restrict__ Kh,
    _Float16* __restrict__ Vh,
    float* __restrict__ Kout, float* __restrict__ Vout)
{
  __shared__ float red[128];
  __shared__ float qs[128];
  __shared__ float ks[128];

  const int d  = threadIdx.x;
  const int l  = blockIdx.x % L_;
  const int bh = blockIdx.x / L_;
  const int h  = bh % H_;
  const int b  = bh / H_;

  const size_t base = ((size_t)(b * L_ + l) * 3 * DIM_) + (size_t)h * D_ + d;
  const float q = qkv[base];
  const float k = qkv[base + DIM_];
  const float v = qkv[base + 2 * DIM_];

  red[d] = q * q;
  __syncthreads();
  for (int s = 64; s > 0; s >>= 1) {
    if (d < s) red[d] += red[d + s];
    __syncthreads();
  }
  const float qrs = rsqrtf(fmaxf(red[0] * (1.0f / D_), EPS_));
  __syncthreads();

  red[d] = k * k;
  __syncthreads();
  for (int s = 64; s > 0; s >>= 1) {
    if (d < s) red[d] += red[d + s];
    __syncthreads();
  }
  const float krs = rsqrtf(fmaxf(red[0] * (1.0f / D_), EPS_));
  __syncthreads();

  const float qn = qw[d] * q * qrs;
  const float kn = kw[d] * k * krs;
  qs[d] = qn;
  ks[d] = kn;
  __syncthreads();

  const float qrot = (d < 64) ? -qs[d + 64] : qs[d - 64];
  const float krot = (d < 64) ? -ks[d + 64] : ks[d - 64];
  const float cv = cosT[l * D_ + d];
  const float sv = sinT[l * D_ + d];

  const float qr = qn * cv + qrot * sv;
  const float kr = kn * cv + krot * sv;

  const size_t oi = ((size_t)bh * L_ + l) * D_ + d;
  Qh[oi]   = (_Float16)qr;
  Kh[oi]   = (_Float16)kr;
  Vh[oi]   = (_Float16)v;
  Kout[oi] = kr;
  Vout[oi] = v;
}

// ---------------------------------------------------------------------------
// Causal flash attention with tanh softcap.
// Grid: (L/64 query tiles, B*H heads). 256 threads = 8 waves.
// K/V tiles staged f16 global -> LDS via global_load_async_to_lds_b128.
// Output written f16 (feeds the f16 out-projection GEMM).
// ---------------------------------------------------------------------------
__global__ __launch_bounds__(256) void attn_fwd(
    const _Float16* __restrict__ Q, const _Float16* __restrict__ Kt,
    const _Float16* __restrict__ Vt, _Float16* __restrict__ O)
{
  constexpr int LDK = 136; // 128 + 8 halves; 272 B row stride (16B multiple)
  constexpr int LDS = 72;  // 64 + 8
  __shared__ __align__(16) _Float16 Ksh[64 * LDK];
  __shared__ __align__(16) _Float16 Vsh[64 * LDK];
  __shared__ float    Ssh[64 * LDS];
  __shared__ _Float16 Psh[64 * LDS];
  __shared__ float row_m[64], row_l[64], row_s[64];

  const int tid  = threadIdx.x;
  const int lane = tid & 31;
  const int w    = tid >> 5;
  const int r16  = lane & 15;
  const int hi   = lane >> 4;

  const int qt    = blockIdx.x;
  const int bh    = blockIdx.y;
  const int qbase = qt * 64;

  const _Float16* Qp = Q  + (size_t)bh * L_ * D_;
  const _Float16* Kp = Kt + (size_t)bh * L_ * D_;
  const _Float16* Vp = Vt + (size_t)bh * L_ * D_;

  const int mt = w & 3;
  const int g  = w >> 2;

  // Q fragments held in registers: 4 K-steps of 32 covering D=128
  v16h qf[4];
  {
    const _Float16* qrow = Qp + (size_t)(qbase + mt * 16 + r16) * D_;
#pragma unroll
    for (int kk = 0; kk < 4; ++kk)
#pragma unroll
      for (int j = 0; j < 16; ++j)
        qf[kk][j] = qrow[kk * 32 + kmap16(j, hi)];
  }

  v8f oacc[4] = {};
  if (tid < 64) { row_m[tid] = -1e30f; row_l[tid] = 0.0f; }
  __syncthreads();

  // staging geometry: row = tid/4, cols (tid%4)*32 .. +31 (halves)
  const int srr = tid >> 2;
  const int sc0 = (tid & 3) * 32;

  for (int kt = 0; kt <= qt; ++kt) {
    const int kb = kt * 64;

    // async-stage K,V tile (64 x 128 f16): 4x b128 per thread per tensor
    {
      const _Float16* kp = Kp + (size_t)(kb + srr) * D_ + sc0;
      const _Float16* vp = Vp + (size_t)(kb + srr) * D_ + sc0;
      const unsigned kdst = lds_off(&Ksh[srr * LDK + sc0]);
      const unsigned vdst = lds_off(&Vsh[srr * LDK + sc0]);
#pragma unroll
      for (int i = 0; i < 4; ++i) {
        async_copy_b128(kdst + i * 16, kp + i * 8);
        async_copy_b128(vdst + i * 16, vp + i * 8);
      }
      wait_asynccnt0();
    }
    __syncthreads();

    // S = Q . K^T
#pragma unroll
    for (int nn = 0; nn < 2; ++nn) {
      const int nt = g * 2 + nn;
      v8f sacc = {};
#pragma unroll
      for (int kk = 0; kk < 4; ++kk) {
        v16h bfrag;
#pragma unroll
        for (int j = 0; j < 16; ++j)
          bfrag[j] = Ksh[(nt * 16 + r16) * LDK + kk * 32 + kmap16(j, hi)];
        sacc = wmma_f16(qf[kk], bfrag, sacc);
      }
#pragma unroll
      for (int r = 0; r < 8; ++r)
        Ssh[(mt * 16 + hi * 8 + r) * LDS + nt * 16 + r16] = sacc[r];
    }
    __syncthreads();

    // online softmax with softcap + causal mask (threads 0..63, one per q row)
    if (tid < 64) {
      const int qrow = qbase + tid;
      const float mold = row_m[tid];
      float mx = mold;
#pragma unroll 8
      for (int jj = 0; jj < 64; ++jj) {
        float sv = Ssh[tid * LDS + jj] * SCALE_;
        sv = tanhf(sv * (1.0f / CAP_)) * CAP_;
        if (kb + jj > qrow) sv = -1e30f;
        Ssh[tid * LDS + jj] = sv;
        mx = fmaxf(mx, sv);
      }
      const float alpha = __expf(mold - mx);
      float sum = 0.0f;
#pragma unroll 8
      for (int jj = 0; jj < 64; ++jj) {
        const float p = __expf(Ssh[tid * LDS + jj] - mx);
        sum += p;
        Psh[tid * LDS + jj] = (_Float16)p;
      }
      row_l[tid] = row_l[tid] * alpha + sum;
      row_m[tid] = mx;
      row_s[tid] = alpha;
    }
    __syncthreads();

    // O = O*alpha + P . V
    v16h pa[2];
#pragma unroll
    for (int kk = 0; kk < 2; ++kk)
#pragma unroll
      for (int j = 0; j < 16; ++j)
        pa[kk][j] = Psh[(mt * 16 + r16) * LDS + kk * 32 + kmap16(j, hi)];

#pragma unroll
    for (int dt = 0; dt < 4; ++dt) {
      const int dtg = g * 4 + dt;
      v8f acc = oacc[dt];
#pragma unroll
      for (int r = 0; r < 8; ++r)
        acc[r] *= row_s[mt * 16 + hi * 8 + r];
#pragma unroll
      for (int kk = 0; kk < 2; ++kk) {
        v16h vb;
#pragma unroll
        for (int j = 0; j < 16; ++j)
          vb[j] = Vsh[(kk * 32 + kmap16(j, hi)) * LDK + dtg * 16 + r16];
        acc = wmma_f16(pa[kk], vb, acc);
      }
      oacc[dt] = acc;
    }
    __syncthreads();
  }

  // epilogue: O / l, written f16 as [B, L, H*D] for the output projection
  const int b = bh / H_;
  const int h = bh % H_;
#pragma unroll
  for (int dt = 0; dt < 4; ++dt) {
    const int dtg = g * 4 + dt;
#pragma unroll
    for (int r = 0; r < 8; ++r) {
      const int qm  = mt * 16 + hi * 8 + r;
      const float inv = 1.0f / row_l[qm];
      O[((size_t)(b * L_ + qbase + qm)) * DIM_ + h * D_ + dtg * 16 + r16] =
          (_Float16)(oacc[dt][r] * inv);
    }
  }
}

// ---------------------------------------------------------------------------
extern "C" void kernel_launch(void* const* d_in, const int* in_sizes, int n_in,
                              void* d_out, int out_size, void* d_ws, size_t ws_size,
                              hipStream_t stream)
{
  (void)in_sizes; (void)n_in; (void)out_size; (void)ws_size;

  const float* x     = (const float*)d_in[0];
  const float* w_qkv = (const float*)d_in[1];
  const float* w_out = (const float*)d_in[2];
  const float* qw    = (const float*)d_in[3];
  const float* kw    = (const float*)d_in[4];
  const float* cosT  = (const float*)d_in[5];
  const float* sinT  = (const float*)d_in[6];
  // d_in[7] (mask) unused: causal mask applied analytically.

  float* out   = (float*)d_out;                       // [B, L, DIM]
  float* k_out = out + (size_t)B_ * L_ * DIM_;        // [B, H, L, D]
  float* v_out = k_out + (size_t)B_ * H_ * L_ * D_;   // [B, H, L, D]

  const size_t NX    = (size_t)B_ * L_ * DIM_;        // 8,388,608
  const size_t NWQKV = (size_t)3 * DIM_ * DIM_;       // 12,582,912
  const size_t NWOUT = (size_t)DIM_ * DIM_;           // 4,194,304
  const size_t NBHLD = (size_t)B_ * H_ * L_ * D_;     // 8,388,608

  _Float16* xh    = (_Float16*)d_ws;        // 16 MB (reused as Oh after GEMM1)
  _Float16* wqkvh = xh + NX;                // 24 MB
  _Float16* wouth = wqkvh + NWQKV;          // 8 MB
  _Float16* Qh    = wouth + NWOUT;          // 16 MB
  _Float16* Kh    = Qh + NBHLD;             // 16 MB
  _Float16* Vh    = Kh + NBHLD;             // 16 MB
  float*    qkv   = (float*)(Vh + NBHLD);   // 96 MB
  _Float16* Oh    = xh;                     // alias: x dead after GEMM1

  // 0) one-time f32 -> f16 conversion of GEMM operands
  cvt_f32_f16<<<dim3((int)(NX    / 2048)), 256, 0, stream>>>(x,     xh,    (int)NX);
  cvt_f32_f16<<<dim3((int)(NWQKV / 2048)), 256, 0, stream>>>(w_qkv, wqkvh, (int)NWQKV);
  cvt_f32_f16<<<dim3((int)(NWOUT / 2048)), 256, 0, stream>>>(w_out, wouth, (int)NWOUT);

  // 1) qkv = x @ w_qkv^T    [4096 x 6144 x 2048]
  gemm_h<<<dim3(3 * DIM_ / 64, B_ * L_ / 128), 256, 0, stream>>>(
      xh, wqkvh, qkv, B_ * L_, 3 * DIM_, DIM_);

  // 2) RMS-norm + RoPE; f32 K/V -> d_out, f16 Q/K/V -> workspace
  normrope<<<dim3(B_ * H_ * L_), 128, 0, stream>>>(
      qkv, qw, kw, cosT, sinT, Qh, Kh, Vh, k_out, v_out);

  // 3) causal softcapped flash attention (async K/V staging, WMMA)
  attn_fwd<<<dim3(L_ / 64, B_ * H_), 256, 0, stream>>>(
      Qh, Kh, Vh, Oh);

  // 4) out = o @ w_out^T    [4096 x 2048 x 2048]
  gemm_h<<<dim3(DIM_ / 64, B_ * L_ / 128), 256, 0, stream>>>(
      Oh, wouth, out, B_ * L_, DIM_, DIM_);
}
